// TextClassifier_30820685316818
// MI455X (gfx1250) — matrix operand order repaired and checked by reading, not compiled, not run on hardware
//
#include <hip/hip_runtime.h>
#include <hip/hip_bf16.h>
#include <math.h>

// ---------------- WMMA fragment types (wave32, gfx1250) ----------------
typedef __attribute__((ext_vector_type(16))) __bf16 bf16x16;
typedef __attribute__((ext_vector_type(8)))  __bf16 bf16x8;
typedef __attribute__((ext_vector_type(8)))  float  v8f;

#define DEVINL static __device__ __forceinline__

// A fragment, bf16 16x32 (MxK), A row-major [*, lda].
// lane l (l<16): rows m0+l, K = k0+0..15 contiguous
// lane l>=16  : rows m0+(l-16), K = k0+16..31 contiguous
DEVINL bf16x16 load_a_frag(const __bf16* A, int lda, int m0, int k0, int lane) {
  const int row = m0 + (lane & 15);
  const int kb  = k0 + ((lane & 16) ? 16 : 0);
  const __bf16* p = A + (size_t)row * lda + kb;
  bf16x8 lo = *(const bf16x8*)(p);
  bf16x8 hi = *(const bf16x8*)(p + 8);
  bf16x16 r;
#pragma unroll
  for (int i = 0; i < 8; ++i) { r[i] = lo[i]; r[i + 8] = hi[i]; }
  return r;
}

// B fragment, bf16 32x16 (KxN), from TRANSPOSED weights BT[N, K] (ld = ldb).
// lane l: N = n0+(l&15); K = k0+0..15 (l<16) or k0+16..31 (l>=16), contiguous.
DEVINL bf16x16 load_bt_frag(const __bf16* BT, int ldb, int n0, int k0, int lane) {
  const int row = n0 + (lane & 15);
  const int kb  = k0 + ((lane & 16) ? 16 : 0);
  return *(const bf16x16*)(BT + (size_t)row * ldb + kb);
}

// ---------------- weight convert: W[K,4H] f32 -> WT[4*Hp, Kp] bf16 (gate-padded, transposed)
__global__ void convert_wt_kernel(const float* __restrict__ W, __bf16* __restrict__ WT,
                                  int K, int H, int Kp, int Hp) {
  const long long total = 4LL * Hp * Kp;
  long long idx = (long long)blockIdx.x * blockDim.x + threadIdx.x;
  if (idx >= total) return;
  const int k = (int)(idx % Kp);
  const int nPad = (int)(idx / Kp);
  const int g = nPad / Hp, j = nPad - g * Hp;
  float v = 0.0f;
  if (k < K && j < H) v = W[(size_t)k * (4 * H) + g * H + j];
  WT[idx] = (__bf16)v;
}

// ---------------- embedding gather -> bf16 padded [SLB, Ep]
__global__ void embed_kernel(const int* __restrict__ x, const float* __restrict__ emb,
                             __bf16* __restrict__ out, int SLB, int E, int Ep) {
  long long idx = (long long)blockIdx.x * blockDim.x + threadIdx.x;
  if (idx >= (long long)SLB * Ep) return;
  const int k = (int)(idx % Ep);
  const int r = (int)(idx / Ep);
  float v = 0.0f;
  if (k < E) v = emb[(size_t)x[r] * E + k];
  out[idx] = (__bf16)v;
}

// ---------------- big parallel GEMM: C[M, Np] f32 = A[M, Kp] bf16 @ BT^T + bias
// 4 waves/block; each wave: one 16-row M tile x four 16-col N tiles (A reuse).
__global__ void __launch_bounds__(128)
gemm_bf16_bias(const __bf16* __restrict__ A, const __bf16* __restrict__ BT,
               const float* __restrict__ bias, float* __restrict__ C,
               int Kp, int Np, int H, int Hp) {
  const int lane  = threadIdx.x & 31;
  const int waveG = blockIdx.x * 4 + (threadIdx.x >> 5);
  const int nGrp  = Np >> 6;
  const int m0    = (waveG / nGrp) << 4;
  const int n0    = (waveG % nGrp) << 6;
  v8f acc[4] = {};
  for (int k0 = 0; k0 < Kp; k0 += 32) {
    bf16x16 a = load_a_frag(A, Kp, m0, k0, lane);
#pragma unroll
    for (int t = 0; t < 4; ++t) {
      bf16x16 b = load_bt_frag(BT, Kp, n0 + t * 16, k0, lane);
      acc[t] = __builtin_amdgcn_wmma_f32_16x16x32_bf16(false, a, false, b,
                                                       (short)0, acc[t], false, false);
    }
  }
  const int mb = m0 + ((lane & 16) ? 8 : 0);
#pragma unroll
  for (int t = 0; t < 4; ++t) {
    const int n = n0 + t * 16 + (lane & 15);
    const int g = n / Hp, j = n - g * Hp;
    const float bv = (j < H) ? bias[g * H + j] : 0.0f;
#pragma unroll
    for (int v = 0; v < 8; ++v)
      C[(size_t)(mb + v) * Np + n] = acc[t][v] + bv;
  }
}

// ---------------- one LSTM timestep (serial part), fully fused:
// block = 128 thr = 4 waves (16 batch rows each); blockIdx.x = 16-col hidden tile.
// Each wave accumulates the 4 gate tiles sharing one A fragment (h_{t-1}).
__global__ void __launch_bounds__(128)
lstm_step_kernel(const __bf16* __restrict__ hprev, const __bf16* __restrict__ WhT,
                 const float* __restrict__ Gx, float* __restrict__ c,
                 __bf16* __restrict__ hout, float* __restrict__ yout,
                 int Hp, int H) {
  const int lane = threadIdx.x & 31;
  const int m0   = (threadIdx.x >> 5) << 4;
  const int j0   = blockIdx.x << 4;
  v8f acc[4] = {};
  for (int k0 = 0; k0 < Hp; k0 += 32) {
    bf16x16 a = load_a_frag(hprev, Hp, m0, k0, lane);
#pragma unroll
    for (int g = 0; g < 4; ++g) {
      bf16x16 b = load_bt_frag(WhT, Hp, g * Hp + j0, k0, lane);
      acc[g] = __builtin_amdgcn_wmma_f32_16x16x32_bf16(false, a, false, b,
                                                       (short)0, acc[g], false, false);
    }
  }
  const int n  = j0 + (lane & 15);
  const int mb = m0 + ((lane & 16) ? 8 : 0);
  const bool valid = (n < H);
  const int Np = 4 * Hp;
#pragma unroll
  for (int v = 0; v < 8; ++v) {
    const int m = mb + v;
    const float gi = acc[0][v] + Gx[(size_t)m * Np + 0 * Hp + n];
    const float gf = acc[1][v] + Gx[(size_t)m * Np + 1 * Hp + n];
    const float gg = acc[2][v] + Gx[(size_t)m * Np + 2 * Hp + n];
    const float go = acc[3][v] + Gx[(size_t)m * Np + 3 * Hp + n];
    float cn = 0.0f, h = 0.0f;
    if (valid) {
      const float cold = c[(size_t)m * Hp + n];
      const float si = 1.0f / (1.0f + __expf(-gi));
      const float sf = 1.0f / (1.0f + __expf(-gf));
      const float so = 1.0f / (1.0f + __expf(-go));
      cn = sf * cold + si * tanhf(gg);
      h  = so * tanhf(cn);
    }
    c[(size_t)m * Hp + n]    = cn;               // keep pad region at exact 0
    hout[(size_t)m * Hp + n] = (__bf16)h;
    if (yout != nullptr && valid) yout[(size_t)m * H + n] = h;
  }
}

// ---------------- pooling: [last_h | maxpool | meanpool] over time
__global__ void pool_kernel(const float* __restrict__ y3, float* __restrict__ pooled,
                            int sl, int bs, int E) {
  int idx = blockIdx.x * blockDim.x + threadIdx.x;
  if (idx >= bs * E) return;
  const int b = idx / E, k = idx - b * E;
  float mx = -INFINITY, sm = 0.0f;
  for (int t = 0; t < sl; ++t) {
    const float v = y3[((size_t)t * bs + b) * E + k];
    mx = fmaxf(mx, v); sm += v;
  }
  pooled[(size_t)b * 3 * E + k]         = y3[((size_t)(sl - 1) * bs + b) * E + k];
  pooled[(size_t)b * 3 * E + E + k]     = mx;
  pooled[(size_t)b * 3 * E + 2 * E + k] = sm / (float)sl;
}

// ---------------- per-(i,b) head: score[i*bs+b, c] (tiny; scalar path is fine)
__global__ void __launch_bounds__(64)
head_kernel(const float* __restrict__ pooled, const int* __restrict__ domain,
            const float* __restrict__ HW1, const float* __restrict__ Hb1,
            const float* __restrict__ HW2, const float* __restrict__ Hb2,
            float* __restrict__ score, int bs, int F, int H2, int ncls) {
  const int i = blockIdx.x / bs, b = blockIdx.x - i * bs;
  const int d = domain[i];
  __shared__ float hsh[64];
  const int j = threadIdx.x;
  float hv = 0.0f;
  if (j < H2) {
    float s = Hb1[(size_t)d * H2 + j];
    const float* w = HW1 + (size_t)d * F * H2 + j;
    const float* p = pooled + (size_t)b * F;
    for (int f = 0; f < F; ++f) s += p[f] * w[(size_t)f * H2];
    hv = fmaxf(s, 0.0f);
  }
  hsh[j] = hv;
  __syncthreads();
  if (j < ncls) {
    float s = Hb2[(size_t)d * ncls + j];
    for (int q = 0; q < H2; ++q) s += hsh[q] * HW2[((size_t)d * H2 + q) * ncls + j];
    score[(size_t)(i * bs + b) * ncls + j] = s;
  }
}

// =======================================================================
extern "C" void kernel_launch(void* const* d_in, const int* in_sizes, int n_in,
                              void* d_out, int out_size, void* d_ws, size_t ws_size,
                              hipStream_t stream) {
  (void)in_sizes; (void)n_in; (void)out_size;
  const int SL = 512, BS = 64, SLB = SL * BS;
  const int E = 400, Ep = 416;          // K-pad to mult of 32, N-pad to mult of 16
  const int H = 1150, Hp = 1152;
  const int H3 = 400, Hp3 = 416;        // layer-3 hidden (EMB), padded
  const int HID2 = 50, NCLS = 2, F = 3 * E;

  const int*   x   = (const int*)d_in[0];
  const int*   dom = (const int*)d_in[1];
  const float* emb = (const float*)d_in[2];
  const float* Wx1 = (const float*)d_in[3];
  const float* Wh1 = (const float*)d_in[4];
  const float* b1  = (const float*)d_in[5];
  const float* Wx2 = (const float*)d_in[6];
  const float* Wh2 = (const float*)d_in[7];
  const float* b2  = (const float*)d_in[8];
  const float* Wx3 = (const float*)d_in[9];
  const float* Wh3 = (const float*)d_in[10];
  const float* b3  = (const float*)d_in[11];
  const float* HW1 = (const float*)d_in[12];
  const float* Hb1 = (const float*)d_in[13];
  const float* HW2 = (const float*)d_in[14];
  const float* Hb2 = (const float*)d_in[15];

  float* score = (float*)d_out;                              // [bs*bs, 2]
  float* y3    = (float*)d_out + (size_t)BS * BS * NCLS;     // [sl, bs, 400] fp32

  // ---- workspace carve-out (256B aligned)
  char* wsb = (char*)d_ws;
  size_t off = 0;
  auto alloc = [&](size_t bytes) -> char* {
    char* p = wsb + off;
    off = (off + bytes + 255) & ~(size_t)255;
    return p;
  };
  __bf16* Wx1T = (__bf16*)alloc((size_t)4 * Hp  * Ep * 2);
  __bf16* Wh1T = (__bf16*)alloc((size_t)4 * Hp  * Hp * 2);
  __bf16* Wx2T = (__bf16*)alloc((size_t)4 * Hp  * Hp * 2);
  __bf16* Wh2T = (__bf16*)alloc((size_t)4 * Hp  * Hp * 2);
  __bf16* Wx3T = (__bf16*)alloc((size_t)4 * Hp3 * Hp * 2);
  __bf16* Wh3T = (__bf16*)alloc((size_t)4 * Hp3 * Hp3 * 2);
  __bf16* Xbf0 = (__bf16*)alloc((size_t)SLB * Ep * 2);
  __bf16* Ybf1 = (__bf16*)alloc((size_t)(SL + 1) * BS * Hp * 2);   // slice 0 = h0 = 0
  __bf16* Ybf2 = (__bf16*)alloc((size_t)(SL + 1) * BS * Hp * 2);
  __bf16* Ybf3 = (__bf16*)alloc((size_t)(SL + 1) * BS * Hp3 * 2);
  float*  Gx   = (float*)alloc((size_t)SLB * 4 * Hp * 4);          // reused per layer
  float*  c1   = (float*)alloc((size_t)BS * Hp * 4);
  float*  c2   = (float*)alloc((size_t)BS * Hp * 4);
  float*  c3   = (float*)alloc((size_t)BS * Hp3 * 4);
  float*  pooled = (float*)alloc((size_t)BS * F * 4);
  if (off > ws_size) return;  // workspace too small: bail deterministically

  auto g256 = [](long long n) { return (unsigned)((n + 255) / 256); };

  // 1) weight conversion (f32 -> transposed gate-padded bf16)
  convert_wt_kernel<<<g256(4LL * Hp * Ep), 256, 0, stream>>>(Wx1, Wx1T, E,  H,  Ep, Hp);
  convert_wt_kernel<<<g256(4LL * Hp * Hp), 256, 0, stream>>>(Wh1, Wh1T, H,  H,  Hp, Hp);
  convert_wt_kernel<<<g256(4LL * Hp * Hp), 256, 0, stream>>>(Wx2, Wx2T, H,  H,  Hp, Hp);
  convert_wt_kernel<<<g256(4LL * Hp * Hp), 256, 0, stream>>>(Wh2, Wh2T, H,  H,  Hp, Hp);
  convert_wt_kernel<<<g256(4LL * Hp3 * Hp), 256, 0, stream>>>(Wx3, Wx3T, H,  H3, Hp, Hp3);
  convert_wt_kernel<<<g256(4LL * Hp3 * Hp3), 256, 0, stream>>>(Wh3, Wh3T, H3, H3, Hp3, Hp3);

  // 2) embedding gather -> bf16 padded
  embed_kernel<<<g256((long long)SLB * Ep), 256, 0, stream>>>(x, emb, Xbf0, SLB, E, Ep);

  // 3) zero h0 slices and cell states (capture-safe stream memsets)
  hipMemsetAsync(Ybf1, 0, (size_t)BS * Hp  * 2, stream);
  hipMemsetAsync(Ybf2, 0, (size_t)BS * Hp  * 2, stream);
  hipMemsetAsync(Ybf3, 0, (size_t)BS * Hp3 * 2, stream);
  hipMemsetAsync(c1, 0, (size_t)BS * Hp  * 4, stream);
  hipMemsetAsync(c2, 0, (size_t)BS * Hp  * 4, stream);
  hipMemsetAsync(c3, 0, (size_t)BS * Hp3 * 4, stream);

  // ---- Layer 1 ----
  {
    const int Np = 4 * Hp;
    const unsigned blocks = (unsigned)((SLB / 16) * (Np / 64) / 4);
    gemm_bf16_bias<<<blocks, 128, 0, stream>>>(Xbf0, Wx1T, b1, Gx, Ep, Np, H, Hp);
    for (int t = 0; t < SL; ++t)
      lstm_step_kernel<<<Hp / 16, 128, 0, stream>>>(
          Ybf1 + (size_t)t * BS * Hp, Wh1T, Gx + (size_t)t * BS * Np, c1,
          Ybf1 + (size_t)(t + 1) * BS * Hp, nullptr, Hp, H);
  }
  // ---- Layer 2 ---- (input = Ybf1 slices 1..SL, contiguous [SLB, Hp])
  {
    const int Np = 4 * Hp;
    const unsigned blocks = (unsigned)((SLB / 16) * (Np / 64) / 4);
    gemm_bf16_bias<<<blocks, 128, 0, stream>>>(Ybf1 + (size_t)BS * Hp, Wx2T, b2, Gx, Hp, Np, H, Hp);
    for (int t = 0; t < SL; ++t)
      lstm_step_kernel<<<Hp / 16, 128, 0, stream>>>(
          Ybf2 + (size_t)t * BS * Hp, Wh2T, Gx + (size_t)t * BS * Np, c2,
          Ybf2 + (size_t)(t + 1) * BS * Hp, nullptr, Hp, H);
  }
  // ---- Layer 3 ---- (writes fp32 y3 straight into d_out)
  {
    const int Np = 4 * Hp3;
    const unsigned blocks = (unsigned)((SLB / 16) * (Np / 64) / 4);
    gemm_bf16_bias<<<blocks, 128, 0, stream>>>(Ybf2 + (size_t)BS * Hp, Wx3T, b3, Gx, Hp, Np, H3, Hp3);
    for (int t = 0; t < SL; ++t)
      lstm_step_kernel<<<Hp3 / 16, 128, 0, stream>>>(
          Ybf3 + (size_t)t * BS * Hp3, Wh3T, Gx + (size_t)t * BS * Np, c3,
          Ybf3 + (size_t)(t + 1) * BS * Hp3, y3 + (size_t)t * BS * E, Hp3, H3);
  }

  // ---- pooling + per-sample-domain heads ----
  pool_kernel<<<g256((long long)BS * E), 256, 0, stream>>>(y3, pooled, SL, BS, E);
  head_kernel<<<BS * BS, 64, 0, stream>>>(pooled, dom, HW1, Hb1, HW2, Hb2,
                                          score, BS, F, HID2, NCLS);
}
